// ScaledLinear_13700945674529
// MI455X (gfx1250) — compile-verified, matrix operand order checked by reading, not run on hardware
//
#include <hip/hip_runtime.h>

typedef __attribute__((ext_vector_type(16))) int   v16i;
typedef __attribute__((ext_vector_type(8)))  float v8f;

#define FP8MAX 448.0f

// ---- CDNA5 async global->LDS path (guarded; falls back to sync staging) ----
#if __has_builtin(__builtin_amdgcn_global_load_async_to_lds_b128)
#define USE_ASYNC_LDS 1
typedef int v4i_ls __attribute__((vector_size(16)));   // type per hipcc diagnostic
__device__ inline void async_cp16(const unsigned char* g, unsigned char* l) {
  __builtin_amdgcn_global_load_async_to_lds_b128(
      (v4i_ls*)g, (v4i_ls*)l, /*offset=*/0, /*cpol=*/0);
}
__device__ inline void wait_async0() {
#if __has_builtin(__builtin_amdgcn_s_wait_asynccnt)
  __builtin_amdgcn_s_wait_asynccnt(0);
#else
  asm volatile("s_wait_asynccnt 0x0" ::: "memory");
#endif
}
#endif

// ---------------------------------------------------------------- init
__global__ void ScaledLinear_init_hdr(unsigned* hdr) {
  if (threadIdx.x == 0) { hdr[0] = 0u; hdr[1] = 0u; }
}

// ---------------------------------------------------------------- amax
__global__ __launch_bounds__(256) void ScaledLinear_amax(
    const float* __restrict__ x, long n4, unsigned* __restrict__ out) {
  __shared__ float sm[256];
  float m = 0.0f;
  const float4* p = (const float4*)x;
  for (long i = (long)blockIdx.x * blockDim.x + threadIdx.x; i < n4;
       i += (long)gridDim.x * blockDim.x) {
    float4 v = p[i];
    m = fmaxf(m, fmaxf(fmaxf(fabsf(v.x), fabsf(v.y)),
                       fmaxf(fabsf(v.z), fabsf(v.w))));
  }
  sm[threadIdx.x] = m;
  __syncthreads();
  for (int s = 128; s > 0; s >>= 1) {
    if ((int)threadIdx.x < s)
      sm[threadIdx.x] = fmaxf(sm[threadIdx.x], sm[threadIdx.x + s]);
    __syncthreads();
  }
  if (threadIdx.x == 0) atomicMax(out, __float_as_uint(sm[0]));
}

// ---------------------------------------------------------------- fp8 convert
__device__ inline unsigned char f32_to_e4m3_sw(float f) {
  unsigned u = __float_as_uint(f);
  unsigned s = (u >> 24) & 0x80u;
  unsigned a = u & 0x7FFFFFFFu;
  if (a >= 0x7F800000u) return (unsigned char)(s | 0x7Fu);      // NaN/Inf
  float af = __uint_as_float(a);
  if (af > FP8MAX) af = FP8MAX;                                  // saturate
  a = __float_as_uint(af);
  int e = (int)(a >> 23) - 127;
  unsigned char r;
  if (e < -6) {                                                  // denormal
    float t = af * 512.0f + 12582912.0f;                         // RNE magic
    r = (unsigned char)(__float_as_uint(t) & 0xFFu);
  } else {
    unsigned mant = a & 0x7FFFFFu;
    unsigned keep = mant >> 20;
    unsigned rest = mant & 0xFFFFFu;
    keep += (rest > 0x80000u) || (rest == 0x80000u && (keep & 1u));
    if (keep == 8u) { keep = 0u; e += 1; }
    r = (e > 8) ? (unsigned char)0x7E
                : (unsigned char)(((e + 7) << 3) | keep);
  }
  return (unsigned char)(s | r);
}

__device__ inline unsigned pack4_e4m3(float4 v, float r) {
#if __has_builtin(__builtin_amdgcn_cvt_pk_fp8_f32)
  int pk = 0;
  pk = __builtin_amdgcn_cvt_pk_fp8_f32(v.x * r, v.y * r, pk, false);
  pk = __builtin_amdgcn_cvt_pk_fp8_f32(v.z * r, v.w * r, pk, true);
  return (unsigned)pk;
#else
  return  (unsigned)f32_to_e4m3_sw(v.x * r)
        | ((unsigned)f32_to_e4m3_sw(v.y * r) << 8)
        | ((unsigned)f32_to_e4m3_sw(v.z * r) << 16)
        | ((unsigned)f32_to_e4m3_sw(v.w * r) << 24);
#endif
}

__global__ __launch_bounds__(256) void ScaledLinear_quant(
    const float* __restrict__ x, unsigned char* __restrict__ q, long n16,
    const unsigned* __restrict__ hdr, int which) {
  long i = (long)blockIdx.x * blockDim.x + threadIdx.x;
  if (i >= n16) return;
  float amax = __uint_as_float(hdr[which]);
  float r = FP8MAX / fmaxf(amax, 1e-12f);
  const float4* p = (const float4*)x + i * 4;
  uint4 o;
  o.x = pack4_e4m3(p[0], r);
  o.y = pack4_e4m3(p[1], r);
  o.z = pack4_e4m3(p[2], r);
  o.w = pack4_e4m3(p[3], r);
  *(uint4*)(q + i * 16) = o;
}

// ---------------------------------------------------------------- fp8 GEMM
// out[M,N] = (A8[M,K] . B8[N,K]^T) * s + bias[N]
#define BM 128
#define BN 64
#define BK 128
#define LDAB 144   // BK + 16: 16B-aligned rows, conflict-free bank spread

template <int ROWS>
__device__ inline void stage_tile(const unsigned char* __restrict__ g, long ldg,
                                  unsigned char* l, int t) {
#pragma unroll
  for (int r = 0; r < ROWS; r += 32) {
    int row = r + (t >> 3);
    int col = (t & 7) << 4;
#ifdef USE_ASYNC_LDS
    async_cp16(g + (long)row * ldg + col, l + row * LDAB + col);
#else
    uint4 v = *(const uint4*)(g + (long)row * ldg + col);
    *(uint4*)(l + row * LDAB + col) = v;
#endif
  }
}

__global__ __launch_bounds__(256) void ScaledLinear_gemm_fp8(
    const unsigned char* __restrict__ A8,
    const unsigned char* __restrict__ B8,
    const float* __restrict__ bias,
    float* __restrict__ out,
    const unsigned* __restrict__ hdr,
    int M, int N, int K) {
  __shared__ unsigned char lA[2][BM * LDAB];
  __shared__ unsigned char lB[2][BN * LDAB];

  const int t    = threadIdx.x;
  const int wid  = t >> 5;
  const int lane = t & 31;
  const int hl   = lane >> 4;   // half-wave select
  const int lm   = lane & 15;
  const int wm   = wid >> 1;    // 0..3 : M position within block
  const int wn   = wid & 1;     // 0..1 : N position within block

  const int m0 = blockIdx.y * BM;
  const int n0 = blockIdx.x * BN;

  const unsigned char* gA = A8 + (long)m0 * K;
  const unsigned char* gB = B8 + (long)n0 * K;

  v8f acc[2][2];
#pragma unroll
  for (int i = 0; i < 2; ++i)
#pragma unroll
    for (int j = 0; j < 2; ++j)
#pragma unroll
      for (int r = 0; r < 8; ++r) acc[i][j][r] = 0.0f;

  // prologue: stage first K-chunk into buffer 0
  stage_tile<BM>(gA, K, lA[0], t);
  stage_tile<BN>(gB, K, lB[0], t);

  const int nk = K / BK;
  for (int ib = 0; ib < nk; ++ib) {
    const int cur = ib & 1;
#ifdef USE_ASYNC_LDS
    wait_async0();               // own async writes of lX[cur] have landed
#endif
    __syncthreads();             // all waves' writes visible; prev reads done

    if (ib + 1 < nk) {           // prefetch next chunk into the other buffer
      stage_tile<BM>(gA + (ib + 1) * BK, K, lA[cur ^ 1], t);
      stage_tile<BN>(gB + (ib + 1) * BK, K, lB[cur ^ 1], t);
    }

    // ---- fragments per ISA 8-bit operand layout:
    // lane<16 : K = {0-7,16-23,...}+base ; lane>=16 : +8
    v16i af[2], bf[2];
#pragma unroll
    for (int i = 0; i < 2; ++i) {
      const unsigned long long* pa = (const unsigned long long*)
          (lA[cur] + (wm * 32 + i * 16 + lm) * LDAB + hl * 8);
#pragma unroll
      for (int j = 0; j < 8; ++j) {
        unsigned long long d = pa[2 * j];
        af[i][2 * j]     = (int)(unsigned)d;
        af[i][2 * j + 1] = (int)(unsigned)(d >> 32);
      }
    }
#pragma unroll
    for (int i = 0; i < 2; ++i) {
      const unsigned long long* pb = (const unsigned long long*)
          (lB[cur] + (wn * 32 + i * 16 + lm) * LDAB + hl * 8);
#pragma unroll
      for (int j = 0; j < 8; ++j) {
        unsigned long long d = pb[2 * j];
        bf[i][2 * j]     = (int)(unsigned)d;
        bf[i][2 * j + 1] = (int)(unsigned)(d >> 32);
      }
    }

#pragma unroll
    for (int i = 0; i < 2; ++i)
#pragma unroll
      for (int j = 0; j < 2; ++j)
        acc[i][j] = __builtin_amdgcn_wmma_f32_16x16x128_fp8_fp8(
            af[i], bf[j], (short)0, acc[i][j], false, false);
  }

  // ---- epilogue: dequant scale + bias
  float sx = fmaxf(__uint_as_float(hdr[0]), 1e-12f) * (1.0f / FP8MAX);
  float sw = fmaxf(__uint_as_float(hdr[1]), 1e-12f) * (1.0f / FP8MAX);
  float s = sx * sw;

#pragma unroll
  for (int i = 0; i < 2; ++i)
#pragma unroll
    for (int j = 0; j < 2; ++j) {
      int gm = m0 + wm * 32 + i * 16 + hl * 8;   // C/D: upper half-wave = M+8
      int gn = n0 + wn * 32 + j * 16 + lm;
      float b = bias[gn];
#pragma unroll
      for (int r = 0; r < 8; ++r)
        out[(long)(gm + r) * N + gn] = acc[i][j][r] * s + b;
    }
}

// ---------------------------------------------------------------- launch
extern "C" void kernel_launch(void* const* d_in, const int* in_sizes, int n_in,
                              void* d_out, int out_size, void* d_ws, size_t ws_size,
                              hipStream_t stream) {
  const float* x      = (const float*)d_in[0];
  const float* weight = (const float*)d_in[1];
  const float* bias   = (const float*)d_in[2];
  float* out          = (float*)d_out;

  const int N = in_sizes[2];                 // 4096
  const int K = in_sizes[1] / N;             // 4096
  const int M = in_sizes[0] / K;             // 4096

  unsigned* hdr      = (unsigned*)d_ws;
  unsigned char* X8  = (unsigned char*)d_ws + 256;
  unsigned char* W8  = X8 + (size_t)M * (size_t)K;

  const long nx = (long)M * K;
  const long nw = (long)N * K;

  ScaledLinear_init_hdr<<<1, 64, 0, stream>>>(hdr);

  long nx4 = nx / 4, nw4 = nw / 4;
  int gx = (int)((nx4 + 255) / 256); if (gx > 2048) gx = 2048;
  int gw = (int)((nw4 + 255) / 256); if (gw > 2048) gw = 2048;
  ScaledLinear_amax<<<gx, 256, 0, stream>>>(x, nx4, hdr + 0);
  ScaledLinear_amax<<<gw, 256, 0, stream>>>(weight, nw4, hdr + 1);

  long nx16 = nx / 16, nw16 = nw / 16;
  ScaledLinear_quant<<<(int)((nx16 + 255) / 256), 256, 0, stream>>>(x, X8, nx16, hdr, 0);
  ScaledLinear_quant<<<(int)((nw16 + 255) / 256), 256, 0, stream>>>(weight, W8, nw16, hdr, 1);

  dim3 grid(N / BN, M / BM);
  ScaledLinear_gemm_fp8<<<grid, 256, 0, stream>>>(X8, W8, bias, out, hdr, M, N, K);
}
